// BertAttention_17523466567837
// MI455X (gfx1250) — matrix operand order repaired and checked by reading, not compiled, not run on hardware
//
#include <hip/hip_runtime.h>
#include <hip/hip_bf16.h>
#include <math.h>

// ---------------------------------------------------------------------------
// Types for CDNA5 WMMA (wave32, 16x16x32 bf16 -> f32)
// ---------------------------------------------------------------------------
typedef __attribute__((ext_vector_type(16))) __bf16 v16bf;
typedef __attribute__((ext_vector_type(8)))  __bf16 v8bf;
typedef __attribute__((ext_vector_type(8)))  float  v8f;

#define BN 8
#define SQ 512
#define HD 1024
#define NHEAD 16
#define DH 64

__device__ __forceinline__ v8f wmma_bf16(v16bf a, v16bf b, v8f c) {
  return __builtin_amdgcn_wmma_f32_16x16x32_bf16(
      /*neg_a=*/false, a, /*neg_b=*/false, b,
      /*c_mod=*/(short)0, c, /*reuse_a=*/false, /*reuse_b=*/false);
}

// A fragment (16x32, MxK). Row-major buffer with 'stride' halves per row.
__device__ __forceinline__ v16bf load_a_frag(const __bf16* buf, int row, int stride,
                                             int kk, int lane) {
  int grp = lane >> 4;
  const __bf16* p = buf + (size_t)row * stride + kk;
  v8bf lo = *(const v8bf*)(p + grp * 16);
  v8bf hi = *(const v8bf*)(p + grp * 16 + 8);
  v16bf a;
#pragma unroll
  for (int e = 0; e < 8; ++e) { a[e] = lo[e]; a[8 + e] = hi[e]; }
  return a;
}

// B fragment (32x16, KxN) where B[k,n] = buf[n][kk + k].
__device__ __forceinline__ v16bf load_b_frag(const __bf16* buf, int n, int stride,
                                             int kk, int lane) {
  int grp = lane >> 4;
  return *(const v16bf*)(buf + (size_t)n * stride + kk + grp * 16);
}

// ---------------------------------------------------------------------------
// CDNA5 async memory->LDS (ASYNCcnt-tracked, no VGPR round trip).
// ---------------------------------------------------------------------------
__device__ __forceinline__ void async_b128(unsigned lds_off, unsigned long long gaddr) {
  asm volatile("global_load_async_to_lds_b128 %0, %1, off"
               :: "v"(lds_off), "v"(gaddr) : "memory");
}
__device__ __forceinline__ void wait_async0() {
  asm volatile("s_wait_asynccnt 0x0" ::: "memory");
}
__device__ __forceinline__ void wait_async4() {  // allow the 4 just-issued in flight
  asm volatile("s_wait_asynccnt 0x4" ::: "memory");
}

// Copies 8192 bytes with 128 threads: 4 x b128 per lane.
__device__ __forceinline__ void async_copy_8k(const __bf16* g, __bf16* l, int tid) {
#pragma unroll
  for (int it = 0; it < 4; ++it) {
    unsigned loff = (unsigned)(uintptr_t)l + (unsigned)(tid * 16 + it * 2048);
    unsigned long long ga =
        (unsigned long long)(uintptr_t)g + (unsigned)(tid * 16 + it * 2048);
    async_b128(loff, ga);
  }
}

// ---------------------------------------------------------------------------
// fp32 -> bf16 conversion
// ---------------------------------------------------------------------------
__global__ void cvt_f32_bf16(const float* __restrict__ s, __bf16* __restrict__ d, int n) {
  int i = blockIdx.x * blockDim.x + threadIdx.x;
  if (i < n) d[i] = (__bf16)s[i];
}

// ---------------------------------------------------------------------------
// Generic projection GEMM: out = (X[M,1024] @ W[1024,1024]^T + bias) * scale
// mode 0: head-split bf16 store [B,NH,S,DH]   (M = 4096, m = b*512+s)
// mode 1: pos head-split bf16 store [NH,1024,DH] (M = 1024, m = r)
// mode 2: merged f32 store + residual          (M = 4096)
// Block 256 threads (8 waves), tile 128x128, K step 32.
// Double-buffered async tile staging: 4 b128 per thread per tile-pair.
// ---------------------------------------------------------------------------
__device__ __forceinline__ void gemm_issue_tiles(const __bf16* X, const __bf16* W,
                                                 __bf16* As, __bf16* Bs,
                                                 int m0, int n0, int k0, int tid) {
  const int r  = tid >> 1;
  const unsigned cb = (unsigned)((tid & 1) * 32);  // byte offset within 64B row
  unsigned la = (unsigned)(uintptr_t)As + (unsigned)(r * 64) + cb;
  unsigned lb = (unsigned)(uintptr_t)Bs + (unsigned)(r * 64) + cb;
  unsigned long long gx =
      (unsigned long long)(uintptr_t)(X + (size_t)(m0 + r) * 1024 + k0) + cb;
  unsigned long long gw =
      (unsigned long long)(uintptr_t)(W + (size_t)(n0 + r) * 1024 + k0) + cb;
  async_b128(la,       gx);
  async_b128(la + 16u, gx + 16);
  async_b128(lb,       gw);
  async_b128(lb + 16u, gw + 16);
}

__global__ void gemm_bf16(const __bf16* __restrict__ X, const __bf16* __restrict__ W,
                          const float* __restrict__ bias, const float* __restrict__ residual,
                          float scale, int mode,
                          __bf16* __restrict__ outB, float* __restrict__ outF) {
  __shared__ __bf16 As[2][128 * 32];
  __shared__ __bf16 Bs[2][128 * 32];
  const int tid = threadIdx.x;
  const int lane = tid & 31;
  const int w = tid >> 5;
  const int wm = w >> 1, wn = w & 1;
  const int m0 = blockIdx.y * 128, n0 = blockIdx.x * 128;
  const int grp = lane >> 4, ln = lane & 15;

  v8f acc[2][4];
#pragma unroll
  for (int mi = 0; mi < 2; ++mi)
#pragma unroll
    for (int ni = 0; ni < 4; ++ni) acc[mi][ni] = (v8f)(0.0f);

  gemm_issue_tiles(X, W, As[0], Bs[0], m0, n0, 0, tid);

  for (int k0 = 0; k0 < 1024; k0 += 32) {
    const int cur = (k0 >> 5) & 1;
    if (k0 + 32 < 1024) {
      // fill of As/Bs[cur^1] is safe: last read in the previous iteration,
      // whose trailing barrier all waves have passed.
      gemm_issue_tiles(X, W, As[cur ^ 1], Bs[cur ^ 1], m0, n0, k0 + 32, tid);
      wait_async4();  // older (current-buffer) transfers complete; new 4 in flight
    } else {
      wait_async0();
    }
    __syncthreads();  // current tiles visible block-wide

    v16bf a[2];
#pragma unroll
    for (int mi = 0; mi < 2; ++mi)
      a[mi] = load_a_frag(As[cur], wm * 32 + mi * 16 + ln, 32, 0, lane);
#pragma unroll
    for (int ni = 0; ni < 4; ++ni) {
      v16bf bb = load_b_frag(Bs[cur], wn * 64 + ni * 16 + ln, 32, 0, lane);
#pragma unroll
      for (int mi = 0; mi < 2; ++mi) acc[mi][ni] = wmma_bf16(a[mi], bb, acc[mi][ni]);
    }
    __syncthreads();  // reads done -> buffer may be overwritten next iteration
  }

#pragma unroll
  for (int mi = 0; mi < 2; ++mi) {
#pragma unroll
    for (int ni = 0; ni < 4; ++ni) {
      const int c = n0 + wn * 64 + ni * 16 + ln;
      const float bb = bias ? bias[c] : 0.0f;
#pragma unroll
      for (int g = 0; g < 8; ++g) {
        const int m = m0 + wm * 32 + mi * 16 + grp * 8 + g;
        float v = (acc[mi][ni][g] + bb) * scale;
        if (mode == 2) {
          outF[(size_t)m * 1024 + c] = v + residual[(size_t)m * 1024 + c];
        } else if (mode == 1) {
          const int hh = c >> 6, dd = c & 63;
          outB[((size_t)hh * 1024 + m) * 64 + dd] = (__bf16)v;
        } else {
          const int hh = c >> 6, dd = c & 63;
          const int bi = m >> 9, ss = m & 511;
          outB[((((size_t)bi * NHEAD + hh) * SQ) + ss) * 64 + dd] = (__bf16)v;
        }
      }
    }
  }
}

// ---------------------------------------------------------------------------
// Fused disentangled attention.
// grid = (S/64, NH, B), block = 128 threads (4 waves).
// Dynamic LDS: scores 64x512 f32 (128KB) + Q 8KB + K 8KB + Vt 8KB = 152KB.
// scores[i,j] = q_i.k_j + q_i.posK[i-j+512] + k_j.posQ[i-j+512]
// (q, posQ pre-scaled by 1/sqrt(192); clip is identity for S=SPAN=512)
// ---------------------------------------------------------------------------
__global__ void attn_kernel(const __bf16* __restrict__ Q, const __bf16* __restrict__ K,
                            const __bf16* __restrict__ V, const __bf16* __restrict__ PosK,
                            const __bf16* __restrict__ PosQ, const float* __restrict__ amask,
                            __bf16* __restrict__ ctx) {
  extern __shared__ __align__(128) char smem[];
  float*  sc = (float*)smem;                                   // 64*512 f32
  __bf16* Qs = (__bf16*)(smem + 131072);                       // 64*64 bf16
  __bf16* Ks = (__bf16*)(smem + 131072 + 8192);                // 64*64 bf16 (K tiles)
  __bf16* Vt = (__bf16*)(smem + 131072 + 16384);               // 64*64 bf16, transposed V

  const int i0 = blockIdx.x * 64;
  const int h  = blockIdx.y;
  const int b  = blockIdx.z;
  const int tid = threadIdx.x;
  const int lane = tid & 31;
  const int w = tid >> 5;
  const int grp = lane >> 4, ln = lane & 15;

  const __bf16* Qb = Q + (((size_t)b * NHEAD + h) * SQ) * 64;
  const __bf16* Kb = K + (((size_t)b * NHEAD + h) * SQ) * 64;
  const __bf16* Vb = V + (((size_t)b * NHEAD + h) * SQ) * 64;
  const __bf16* PKh = PosK + (size_t)h * 1024 * 64;
  const __bf16* PQh = PosQ + (size_t)h * 1024 * 64;

  // ---- stage Q tile [64 x 64] via async memory->LDS DMA ----
  async_copy_8k(Qb + (size_t)i0 * 64, Qs, tid);
  {  // prefetch first pos windows while the async copy is in flight
    __builtin_prefetch(PKh + (size_t)(i0 + tid) * 64, 0, 0);
    __builtin_prefetch(PQh + (size_t)(i0 + tid) * 64, 0, 0);
  }
  wait_async0();
  __syncthreads();

  // Q A-fragments for this wave's 16 rows, reused across all tiles.
  v16bf aq[2];
  aq[0] = load_a_frag(Qs, w * 16 + ln, 64, 0, lane);
  aq[1] = load_a_frag(Qs, w * 16 + ln, 64, 32, lane);

  // ---- phase 1: build scores strip ----
  for (int J0 = 0; J0 < SQ; J0 += 64) {
    __syncthreads();
    async_copy_8k(Kb + (size_t)J0 * 64, Ks, tid);
    if (J0 + 64 < SQ) {  // prefetch next tile's pos windows
      const int r0n = i0 - (J0 + 64) + 448;
      __builtin_prefetch(PKh + (size_t)(r0n + tid) * 64, 0, 0);
      __builtin_prefetch(PQh + (size_t)(r0n + tid) * 64, 0, 0);
    }
    wait_async0();
    __syncthreads();

    // (a) content-to-content: Qtile @ Ktile^T
    {
      v8f s4[4];
#pragma unroll
      for (int ni = 0; ni < 4; ++ni) s4[ni] = (v8f)(0.0f);
#pragma unroll
      for (int ni = 0; ni < 4; ++ni) {
        v16bf b0 = load_b_frag(Ks, ni * 16 + ln, 64, 0, lane);
        s4[ni] = wmma_bf16(aq[0], b0, s4[ni]);
        v16bf b1 = load_b_frag(Ks, ni * 16 + ln, 64, 32, lane);
        s4[ni] = wmma_bf16(aq[1], b1, s4[ni]);
      }
#pragma unroll
      for (int ni = 0; ni < 4; ++ni)
#pragma unroll
        for (int g = 0; g < 8; ++g)
          sc[(w * 16 + grp * 8 + g) * SQ + J0 + ni * 16 + ln] = s4[ni][g];
    }
    __syncthreads();

    const int r0 = i0 - J0 + 448;  // window base; always in [0, 896]

    // (b) c2p: tmp[ii,c] = q_ii . posK[r0+c]; scatter jj = ii - c + 64
    {
      v8f t8[8];
#pragma unroll
      for (int ni = 0; ni < 8; ++ni) t8[ni] = (v8f)(0.0f);
#pragma unroll
      for (int ni = 0; ni < 8; ++ni) {
        const int rr = r0 + ni * 16 + ln;
        const __bf16* pb = PKh + (size_t)rr * 64;
        v16bf b0 = *(const v16bf*)(pb + grp * 16);
        t8[ni] = wmma_bf16(aq[0], b0, t8[ni]);
        v16bf b1 = *(const v16bf*)(pb + 32 + grp * 16);
        t8[ni] = wmma_bf16(aq[1], b1, t8[ni]);
      }
#pragma unroll
      for (int ni = 0; ni < 8; ++ni)
#pragma unroll
        for (int g = 0; g < 8; ++g) {
          const int ii = w * 16 + grp * 8 + g;
          const int c  = ni * 16 + ln;
          const int jj = ii - c + 64;
          if (jj >= 0 && jj < 64) sc[ii * SQ + J0 + jj] += t8[ni][g];
        }
    }
    __syncthreads();

    // (c) p2c: tmp[jj,c] = k_jj . posQ[r0+c]; scatter ii = jj + c - 64
    {
      v16bf ak[2];
      ak[0] = load_a_frag(Ks, w * 16 + ln, 64, 0, lane);
      ak[1] = load_a_frag(Ks, w * 16 + ln, 64, 32, lane);
      v8f t8[8];
#pragma unroll
      for (int ni = 0; ni < 8; ++ni) t8[ni] = (v8f)(0.0f);
#pragma unroll
      for (int ni = 0; ni < 8; ++ni) {
        const int rr = r0 + ni * 16 + ln;
        const __bf16* pb = PQh + (size_t)rr * 64;
        v16bf b0 = *(const v16bf*)(pb + grp * 16);
        t8[ni] = wmma_bf16(ak[0], b0, t8[ni]);
        v16bf b1 = *(const v16bf*)(pb + 32 + grp * 16);
        t8[ni] = wmma_bf16(ak[1], b1, t8[ni]);
      }
#pragma unroll
      for (int ni = 0; ni < 8; ++ni)
#pragma unroll
        for (int g = 0; g < 8; ++g) {
          const int jj = w * 16 + grp * 8 + g;
          const int c  = ni * 16 + ln;
          const int ii = jj + c - 64;
          if (ii >= 0 && ii < 64) sc[ii * SQ + J0 + jj] += t8[ni][g];
        }
    }
  }
  __syncthreads();

  // ---- phase 2: masked softmax; probs stored bf16 into each row's own slot ----
  for (int rr = 0; rr < 16; ++rr) {
    const int row = w * 16 + rr;
    const float* mrow = amask + ((size_t)b * SQ + (i0 + row)) * SQ;
    float v[16], mk[16];
    float mx = -3.0e38f;
#pragma unroll
    for (int t = 0; t < 16; ++t) {
      const int col = lane + t * 32;
      mk[t] = mrow[col];
      float s = sc[row * SQ + col];
      v[t] = (mk[t] > 0.0f) ? s : -3.0e38f;
      mx = fmaxf(mx, v[t]);
    }
#pragma unroll
    for (int o = 16; o > 0; o >>= 1) mx = fmaxf(mx, __shfl_xor(mx, o, 32));
    float sum = 0.0f;
#pragma unroll
    for (int t = 0; t < 16; ++t) {
      v[t] = (mk[t] > 0.0f) ? __expf(v[t] - mx) : 0.0f;
      sum += v[t];
    }
#pragma unroll
    for (int o = 16; o > 0; o >>= 1) sum += __shfl_xor(sum, o, 32);
    const float inv = (sum > 0.0f) ? 1.0f / sum : 0.0f;
    __bf16* prow = (__bf16*)(sc + (size_t)row * SQ);
#pragma unroll
    for (int t = 0; t < 16; ++t) prow[lane + t * 32] = (__bf16)(v[t] * inv);
  }

  // ---- phase 3: PV: ctx[64,64] = P[64,512] @ V[512,64] ----
  v8f o4[4];
#pragma unroll
  for (int ni = 0; ni < 4; ++ni) o4[ni] = (v8f)(0.0f);

  const __bf16* prow = (const __bf16*)(sc + (size_t)(w * 16 + ln) * SQ);
  for (int k0 = 0; k0 < SQ; k0 += 64) {
    __syncthreads();
    {  // stage V tile transposed: Vt[d][j] = V[k0+j][d]
      const int r = tid >> 1, c0 = (tid & 1) * 32;
      v16bf x0 = *(const v16bf*)&Vb[(size_t)(k0 + r) * 64 + c0];
      v16bf x1 = *(const v16bf*)&Vb[(size_t)(k0 + r) * 64 + c0 + 16];
#pragma unroll
      for (int e = 0; e < 16; ++e) {
        Vt[(c0 + e) * 64 + r]      = x0[e];
        Vt[(c0 + 16 + e) * 64 + r] = x1[e];
      }
    }
    __syncthreads();
#pragma unroll
    for (int kk = 0; kk < 64; kk += 32) {
      v16bf af = load_a_frag(prow, 0, 0, k0 + kk, lane);  // contiguous bf16 probs
#pragma unroll
      for (int ni = 0; ni < 4; ++ni) {
        v16bf bf = load_b_frag(Vt, ni * 16 + ln, 64, kk, lane);  // B[k=j, n=d]
        o4[ni] = wmma_bf16(af, bf, o4[ni]);
      }
    }
  }

#pragma unroll
  for (int ni = 0; ni < 4; ++ni)
#pragma unroll
    for (int g = 0; g < 8; ++g) {
      const int i = i0 + w * 16 + grp * 8 + g;
      const int d = ni * 16 + ln;
      ctx[((size_t)b * SQ + i) * HD + h * 64 + d] = (__bf16)o4[ni][g];
    }
}

// ---------------------------------------------------------------------------
// LayerNorm + output mask: out = LN(pre) * gamma + beta, times mask2d
// ---------------------------------------------------------------------------
__global__ void ln_kernel(const float* __restrict__ pre, const float* __restrict__ gamma,
                          const float* __restrict__ beta, const float* __restrict__ amask,
                          float* __restrict__ out) {
  const int row = blockIdx.x;
  const int tid = threadIdx.x;
  const float* x = pre + (size_t)row * HD;
  __shared__ float red[256];

  float s = 0.0f;
  for (int c = tid; c < HD; c += 256) s += x[c];
  red[tid] = s; __syncthreads();
  for (int o = 128; o > 0; o >>= 1) { if (tid < o) red[tid] += red[tid + o]; __syncthreads(); }
  const float mu = red[0] * (1.0f / HD);
  __syncthreads();

  float s2 = 0.0f;
  for (int c = tid; c < HD; c += 256) { float d = x[c] - mu; s2 += d * d; }
  red[tid] = s2; __syncthreads();
  for (int o = 128; o > 0; o >>= 1) { if (tid < o) red[tid] += red[tid + o]; __syncthreads(); }
  const float rs = rsqrtf(red[0] * (1.0f / HD) + 1e-7f);

  const int b = row >> 9, ss = row & 511;
  const float f = amask[(size_t)b * SQ * SQ + ss];
  for (int c = tid; c < HD; c += 256)
    out[(size_t)row * HD + c] = ((x[c] - mu) * rs * gamma[c] + beta[c]) * f;
}

// ---------------------------------------------------------------------------
// Host launch
// ---------------------------------------------------------------------------
extern "C" void kernel_launch(void* const* d_in, const int* in_sizes, int n_in,
                              void* d_out, int out_size, void* d_ws, size_t ws_size,
                              hipStream_t stream) {
  const float* hs    = (const float*)d_in[0];
  const float* amask = (const float*)d_in[1];
  const float* rel   = (const float*)d_in[2];
  const float* Wq    = (const float*)d_in[3];
  const float* Wk    = (const float*)d_in[4];
  const float* Wv    = (const float*)d_in[5];
  const float* qb    = (const float*)d_in[6];
  const float* vb    = (const float*)d_in[7];
  const float* Wpk   = (const float*)d_in[8];
  const float* Wpq   = (const float*)d_in[9];
  const float* bpq   = (const float*)d_in[10];
  const float* Wo    = (const float*)d_in[11];
  const float* bo    = (const float*)d_in[12];
  const float* gam   = (const float*)d_in[13];
  const float* bet   = (const float*)d_in[14];
  float* out = (float*)d_out;

  char* w = (char*)d_ws;
  const size_t SZ_HS = (size_t)4096 * 1024 * 2;  // 8 MB
  const size_t SZ_W  = (size_t)1024 * 1024 * 2;  // 2 MB
  __bf16* hs_bf  = (__bf16*)w;               w += SZ_HS;
  __bf16* rel_bf = (__bf16*)w;               w += SZ_W;
  __bf16* Wq_bf  = (__bf16*)w;               w += SZ_W;
  __bf16* Wk_bf  = (__bf16*)w;               w += SZ_W;
  __bf16* Wv_bf  = (__bf16*)w;               w += SZ_W;
  __bf16* Wpk_bf = (__bf16*)w;               w += SZ_W;
  __bf16* Wpq_bf = (__bf16*)w;               w += SZ_W;
  __bf16* Wo_bf  = (__bf16*)w;               w += SZ_W;
  __bf16* Qh     = (__bf16*)w;               w += SZ_HS;
  __bf16* Kh     = (__bf16*)w;               w += SZ_HS;
  __bf16* Vh     = (__bf16*)w;               w += SZ_HS;
  __bf16* PKw    = (__bf16*)w;               w += SZ_W;
  __bf16* PQw    = (__bf16*)w;               w += SZ_W;
  __bf16* ctx    = (__bf16*)w;               w += SZ_HS;
  float*  pre    = (float*)w;                w += (size_t)4096 * 1024 * 4;

  const int NHS = 4096 * 1024, NW = 1024 * 1024;
  cvt_f32_bf16<<<(NHS + 255) / 256, 256, 0, stream>>>(hs, hs_bf, NHS);
  cvt_f32_bf16<<<(NW + 255) / 256, 256, 0, stream>>>(rel, rel_bf, NW);
  cvt_f32_bf16<<<(NW + 255) / 256, 256, 0, stream>>>(Wq, Wq_bf, NW);
  cvt_f32_bf16<<<(NW + 255) / 256, 256, 0, stream>>>(Wk, Wk_bf, NW);
  cvt_f32_bf16<<<(NW + 255) / 256, 256, 0, stream>>>(Wv, Wv_bf, NW);
  cvt_f32_bf16<<<(NW + 255) / 256, 256, 0, stream>>>(Wpk, Wpk_bf, NW);
  cvt_f32_bf16<<<(NW + 255) / 256, 256, 0, stream>>>(Wpq, Wpq_bf, NW);
  cvt_f32_bf16<<<(NW + 255) / 256, 256, 0, stream>>>(Wo, Wo_bf, NW);

  const float inv_scale = 1.0f / sqrtf((float)(DH * 3));  // 1/sqrt(192)

  dim3 gBig(8, 32), gPos(8, 8), blk(256);
  gemm_bf16<<<gBig, blk, 0, stream>>>(hs_bf, Wq_bf, qb, nullptr, inv_scale, 0, Qh, nullptr);
  gemm_bf16<<<gBig, blk, 0, stream>>>(hs_bf, Wk_bf, nullptr, nullptr, 1.0f, 0, Kh, nullptr);
  gemm_bf16<<<gBig, blk, 0, stream>>>(hs_bf, Wv_bf, vb, nullptr, 1.0f, 0, Vh, nullptr);
  gemm_bf16<<<gPos, blk, 0, stream>>>(rel_bf, Wpk_bf, nullptr, nullptr, 1.0f, 1, PKw, nullptr);
  gemm_bf16<<<gPos, blk, 0, stream>>>(rel_bf, Wpq_bf, bpq, nullptr, inv_scale, 1, PQw, nullptr);

  const size_t attn_lds = 131072 + 3 * 8192;  // 155648 bytes
  dim3 gAttn(SQ / 64, NHEAD, BN);
  attn_kernel<<<gAttn, dim3(128), attn_lds, stream>>>(Qh, Kh, Vh, PKw, PQw, amask, ctx);

  gemm_bf16<<<gBig, blk, 0, stream>>>(ctx, Wo_bf, bo, hs, 1.0f, 2, nullptr, pre);

  ln_kernel<<<4096, 256, 0, stream>>>(pre, gam, bet, amask, out);
}
